// AsyncLightBus_9234179686567
// MI455X (gfx1250) — matrix operand order repaired and checked by reading, not compiled, not run on hardware
//
#include <hip/hip_runtime.h>
#include <math.h>

#define BSZ     2
#define SEQ     8192
#define DM      4096
#define BUS     128
#define LCACHE  24
#define LAUG    25

typedef __attribute__((ext_vector_type(8)))  float  v8f;
typedef __attribute__((ext_vector_type(8)))  __bf16 v8bf;
typedef __attribute__((ext_vector_type(16))) __bf16 v16bf;

#if __has_builtin(__builtin_amdgcn_global_load_async_to_lds_b128)
#define ALB_ASYNC_LDS 1
typedef int alb_v4i __attribute__((vector_size(16)));
typedef __attribute__((address_space(1))) alb_v4i* alb_gptr;
typedef __attribute__((address_space(3))) alb_v4i* alb_sptr;
#else
#define ALB_ASYNC_LDS 0
#endif

__device__ __forceinline__ v16bf cat8(v8bf lo, v8bf hi) {
  return __builtin_shufflevector(lo, hi, 0,1,2,3,4,5,6,7,8,9,10,11,12,13,14,15);
}
__device__ __forceinline__ void cvt4(__bf16* d, float4 v) {
  d[0] = (__bf16)v.x; d[1] = (__bf16)v.y; d[2] = (__bf16)v.z; d[3] = (__bf16)v.w;
}

// ---------------- Stage 1: column means of x over S (two-stage, deterministic) ----
__global__ __launch_bounds__(256) void alb_mean_partial(const float* __restrict__ x,
                                                        float* __restrict__ partial) {
  int blk = blockIdx.x;
  int dch = blk & 15, sch = (blk >> 4) & 31, b = blk >> 9;
  int d = dch * 256 + threadIdx.x;
  const float* p = x + ((long long)b * SEQ + (long long)sch * 256) * DM + d;
  float acc = 0.f;
  for (int i = 0; i < 256; ++i) {
    acc += p[(long long)i * DM];
    if ((i & 7) == 0 && i + 8 < 256) __builtin_prefetch(p + (long long)(i + 8) * DM, 0, 3);
  }
  partial[((long long)(b * 32 + sch)) * DM + d] = acc;
}

__global__ __launch_bounds__(256) void alb_mean_reduce(const float* __restrict__ partial,
                                                       float* __restrict__ mean) {
  int blk = blockIdx.x;
  int dch = blk & 15, b = blk >> 4;
  int d = dch * 256 + threadIdx.x;
  float s = 0.f;
  #pragma unroll
  for (int sch = 0; sch < 32; ++sch) s += partial[((long long)(b * 32 + sch)) * DM + d];
  mean[b * DM + d] = s * (1.0f / SEQ);
}

// ---------------- Stage 2: summary = normalize(mean @ Wp^T); build aug + new_cache --
__global__ __launch_bounds__(128) void alb_summary(const float* __restrict__ mean,
                                                   const float* __restrict__ Wp,
                                                   const float* __restrict__ bc,
                                                   float* __restrict__ aug,
                                                   float* __restrict__ out) {
  __shared__ float red[128];
  int b = blockIdx.x, k = threadIdx.x;
  const float* m = mean + b * DM;
  const float* w = Wp + (long long)k * DM;
  float dot = 0.f;
  for (int d = 0; d < DM; ++d) dot += m[d] * w[d];
  red[k] = dot * dot;
  __syncthreads();
  for (int off = 64; off > 0; off >>= 1) {
    if (k < off) red[k] += red[k + off];
    __syncthreads();
  }
  float nrm = sqrtf(red[0]);
  float sm = dot / fmaxf(nrm, 1e-12f);

  // aug[b] : row 0 = summary, rows 1..24 = bus_cache
  aug[b * (LAUG * BUS) + k] = sm;
  for (int idx = k; idx < LCACHE * BUS; idx += 128)
    aug[b * (LAUG * BUS) + BUS + idx] = bc[b * (LCACHE * BUS) + idx];

  // new_cache tail of d_out : rows 0..23 = bus_cache, row 24 = summary
  float* nc = out + (long long)BSZ * SEQ * DM + (long long)b * (LAUG * BUS);
  for (int idx = k; idx < LCACHE * BUS; idx += 128)
    nc[idx] = bc[b * (LCACHE * BUS) + idx];
  nc[LCACHE * BUS + k] = sm;
}

// ---------------- Stage 3: fused q-GEMM -> attention -> modulation-GEMM + residual ---
// Block = 128 threads (4 wave32), M-tile = 64 rows. bf16 WMMA with f32 accumulate.
// LDS layout (phase-overlapped):
//   phase1: ldsAf [64][36]f32 @0 (9216, async-filled), ldsB [128][40]bf16 @9216 (10240)
//   phase2: ldsQ [64][129]f32 @0 (33024), ldsAug [25][132]bf16 @33024,
//           ldsS [64][28]f32 @39632, ldsG [64][136]bf16 @46800  => 64208 B total
__global__ __launch_bounds__(128) void alb_fused(const float* __restrict__ x,
                                                 const float* __restrict__ Wq,
                                                 const float* __restrict__ Wm,
                                                 const float* __restrict__ aug,
                                                 const float* __restrict__ gate,
                                                 float* __restrict__ out) {
  __shared__ __attribute__((aligned(16))) unsigned char smem[64208];
  float*  ldsAf  = (float*)(smem + 0);        // fp32 x chunk [64][36]
  __bf16* ldsB   = (__bf16*)(smem + 9216);    // bf16 Wq chunk [128][40]
  float*  ldsQ   = (float*)(smem + 0);
  __bf16* ldsAug = (__bf16*)(smem + 33024);
  float*  ldsS   = (float*)(smem + 39632);
  __bf16* ldsG   = (__bf16*)(smem + 46800);

  const int tid  = threadIdx.x;
  const int lane = tid & 31;
  const int wave = tid >> 5;
  const int hi16 = lane >> 4;   // 0 | 1 : which half-wave
  const int l16  = lane & 15;

  const int gr = blockIdx.x * 64;            // global row across batches
  const int b  = gr / SEQ;
  const int s0 = gr % SEQ;
  const float* xb = x + ((long long)b * SEQ + s0) * DM;
  float* ob = out + ((long long)b * SEQ + s0) * DM;

  // stage bf16 aug for this batch (region disjoint from phase-1 buffers)
  for (int idx = tid; idx < LAUG * BUS; idx += 128) {
    int l = idx >> 7, c = idx & 127;
    ldsAug[l * 132 + c] = (__bf16)aug[b * (LAUG * BUS) + idx];
  }

  // ======== GEMM1: q[64x128] = x_tile @ Wq^T, K = 4096 in steps of 32 ========
  v8f acc[8] = {};
  const int mrow = wave * 16;
  const int xrow = tid >> 1, xhalf = tid & 1;      // staging assignment: 2 threads / row

  for (int k0 = 0; k0 < DM; k0 += 32) {
    { // x chunk (fp32) -> ldsAf : async DMA to LDS when available
      const float* gsrc = xb + (long long)xrow * DM + k0 + xhalf * 16;
      float* ldst = ldsAf + xrow * 36 + xhalf * 16;
#if ALB_ASYNC_LDS
      #pragma unroll
      for (int i = 0; i < 4; ++i)
        __builtin_amdgcn_global_load_async_to_lds_b128(
            (alb_gptr)(gsrc + i * 4), (alb_sptr)(ldst + i * 4), 0, 0);
#else
      const float4* src4 = (const float4*)gsrc;
      float4* dst4 = (float4*)ldst;
      #pragma unroll
      for (int i = 0; i < 4; ++i) dst4[i] = src4[i];
#endif
    }
    { // Wq chunk -> ldsB (bf16; converted once, reused by 4 waves)
      const float4* src = (const float4*)(Wq + (long long)tid * DM + k0);
      __bf16* dst = ldsB + tid * 40;
      #pragma unroll
      for (int i = 0; i < 8; ++i) cvt4(dst + i * 4, src[i]);
      if (k0 + 32 < DM) __builtin_prefetch(Wq + (long long)tid * DM + k0 + 32, 0, 3);
    }
#if ALB_ASYNC_LDS
    asm volatile("s_wait_asynccnt 0x0" ::: "memory");
#endif
    __syncthreads();

    // A fragment: fp32 LDS -> bf16 regs; lane<16 K{0..7,16..23}, lane>=16 K{8..15,24..31}
    const int abase = hi16 * 8;
    const float* ap = ldsAf + (mrow + l16) * 36 + abase;
    float4 a0 = *(const float4*)(ap);
    float4 a1 = *(const float4*)(ap + 4);
    float4 a2 = *(const float4*)(ap + 16);
    float4 a3 = *(const float4*)(ap + 20);
    v16bf afrag;
    afrag[0]  = (__bf16)a0.x; afrag[1]  = (__bf16)a0.y; afrag[2]  = (__bf16)a0.z; afrag[3]  = (__bf16)a0.w;
    afrag[4]  = (__bf16)a1.x; afrag[5]  = (__bf16)a1.y; afrag[6]  = (__bf16)a1.z; afrag[7]  = (__bf16)a1.w;
    afrag[8]  = (__bf16)a2.x; afrag[9]  = (__bf16)a2.y; afrag[10] = (__bf16)a2.z; afrag[11] = (__bf16)a2.w;
    afrag[12] = (__bf16)a3.x; afrag[13] = (__bf16)a3.y; afrag[14] = (__bf16)a3.z; afrag[15] = (__bf16)a3.w;

    #pragma unroll
    for (int t = 0; t < 8; ++t) {
      // B fragment: lane<16 K{0..7,16..23}, lane>=16 K{8..15,24..31}, column N = t*16+l16
      const __bf16* bp = ldsB + (t * 16 + l16) * 40 + hi16 * 8;
      v16bf bfrag = cat8(*(const v8bf*)(bp), *(const v8bf*)(bp + 16));
      acc[t] = __builtin_amdgcn_wmma_f32_16x16x32_bf16(false, afrag, false, bfrag,
                                                       (short)0, acc[t], false, false);
    }
    __syncthreads();
  }

  // spill scaled q to LDS: C layout row = mrow + hi16*8 + j, col = t*16 + l16
  const float qscl = 0.088388347648318447f;  // 1/sqrt(128)
  #pragma unroll
  for (int t = 0; t < 8; ++t)
    #pragma unroll
    for (int j = 0; j < 8; ++j)
      ldsQ[(mrow + hi16 * 8 + j) * 129 + (t * 16 + l16)] = acc[t][j] * qscl;
  __syncthreads();

  // ======== attention over 25 cache rows (tiny; fp32 VALU) ========
  for (int idx = tid; idx < 64 * LAUG; idx += 128) {
    int r = idx / LAUG, l = idx - r * LAUG;
    const float* qr = ldsQ + r * 129;
    const __bf16* ar = ldsAug + l * 132;
    float s = 0.f;
    #pragma unroll 8
    for (int c = 0; c < BUS; ++c) s += qr[c] * (float)ar[c];
    ldsS[r * 28 + l] = s;
  }
  __syncthreads();
  if (tid < 64) {  // softmax per row
    float* sr = ldsS + tid * 28;
    float mx = -3.402823466e38f;
    #pragma unroll
    for (int l = 0; l < LAUG; ++l) mx = fmaxf(mx, sr[l]);
    float sum = 0.f;
    #pragma unroll
    for (int l = 0; l < LAUG; ++l) { float e = __expf(sr[l] - mx); sr[l] = e; sum += e; }
    float inv = 1.f / sum;
    #pragma unroll
    for (int l = 0; l < LAUG; ++l) sr[l] *= inv;
  }
  __syncthreads();
  for (int idx = tid; idx < 64 * BUS; idx += 128) {  // gathered = attn @ aug
    int r = idx >> 7, c = idx & 127;
    const float* sr = ldsS + r * 28;
    float g = 0.f;
    #pragma unroll
    for (int l = 0; l < LAUG; ++l) g += sr[l] * (float)ldsAug[l * 132 + c];
    ldsG[r * 136 + c] = (__bf16)g;
  }
  __syncthreads();

  // ======== GEMM2: out = x + sigmoid(gate) * (gathered @ Wm^T), K=128, N=4096 ========
  v16bf gf[4];
  {
    const __bf16* grow = ldsG + (mrow + l16) * 136;
    const int abase = hi16 * 16;
    #pragma unroll
    for (int kk = 0; kk < 4; ++kk)
      gf[kk] = cat8(*(const v8bf*)(grow + kk * 32 + abase),
                    *(const v8bf*)(grow + kk * 32 + abase + 8));
  }
  const float gsig = 1.f / (1.f + __expf(-gate[0]));

  for (int i = 0; i < 64; ++i) {
    int n0 = (wave + 4 * i) * 16;
    const float* wrow = Wm + (long long)(n0 + l16) * BUS + hi16 * 16;  // Wm row-major [D][BUS]
    v8f c = {};
    #pragma unroll
    for (int kk = 0; kk < 4; ++kk) {
      const float4* wp = (const float4*)(wrow + kk * 32);
      float4 w0 = wp[0], w1 = wp[1], w2 = wp[2], w3 = wp[3];
      v16bf bfrag;
      bfrag[0]  = (__bf16)w0.x; bfrag[1]  = (__bf16)w0.y; bfrag[2]  = (__bf16)w0.z; bfrag[3]  = (__bf16)w0.w;
      bfrag[4]  = (__bf16)w1.x; bfrag[5]  = (__bf16)w1.y; bfrag[6]  = (__bf16)w1.z; bfrag[7]  = (__bf16)w1.w;
      bfrag[8]  = (__bf16)w2.x; bfrag[9]  = (__bf16)w2.y; bfrag[10] = (__bf16)w2.z; bfrag[11] = (__bf16)w2.w;
      bfrag[12] = (__bf16)w3.x; bfrag[13] = (__bf16)w3.y; bfrag[14] = (__bf16)w3.z; bfrag[15] = (__bf16)w3.w;
      c = __builtin_amdgcn_wmma_f32_16x16x32_bf16(false, gf[kk], false, bfrag,
                                                  (short)0, c, false, false);
    }
    #pragma unroll
    for (int j = 0; j < 8; ++j) {
      int off = (mrow + hi16 * 8 + j) * DM + n0 + l16;
      ob[off] = xb[off] + c[j] * gsig;
    }
  }
}

extern "C" void kernel_launch(void* const* d_in, const int* in_sizes, int n_in,
                              void* d_out, int out_size, void* d_ws, size_t ws_size,
                              hipStream_t stream) {
  const float* x    = (const float*)d_in[0];
  const float* bc   = (const float*)d_in[1];
  const float* Wp   = (const float*)d_in[2];
  const float* Wq   = (const float*)d_in[3];
  const float* Wm   = (const float*)d_in[4];
  const float* gate = (const float*)d_in[5];
  float* out = (float*)d_out;

  float* ws      = (float*)d_ws;
  float* partial = ws;                                  // BSZ*32*DM   = 262144 floats
  float* mean    = partial + (long long)BSZ * 32 * DM;  // BSZ*DM      = 8192 floats
  float* aug     = mean + (long long)BSZ * DM;          // BSZ*LAUG*BUS = 6400 floats

  alb_mean_partial<<<dim3(BSZ * 32 * 16),    dim3(256), 0, stream>>>(x, partial);
  alb_mean_reduce <<<dim3(BSZ * 16),         dim3(256), 0, stream>>>(partial, mean);
  alb_summary     <<<dim3(BSZ),              dim3(128), 0, stream>>>(mean, Wp, bc, aug, out);
  alb_fused       <<<dim3((BSZ * SEQ) / 64), dim3(128), 0, stream>>>(x, Wq, Wm, aug, gate, out);
}
